// HeteroGNN2_39376260169770
// MI455X (gfx1250) — matrix operand order repaired and checked by reading, not compiled, not run on hardware
//
#include <hip/hip_runtime.h>

#define H       128
#define BATCH   64
#define NFACT   100000
#define NCOMP   5000
#define NEDGE   500000
#define DFACT   384
#define DCOMP   64
#define NLAYERS 2

typedef __attribute__((ext_vector_type(16))) _Float16 v16h;
typedef __attribute__((ext_vector_type(8)))  float    v8f;
typedef __attribute__((ext_vector_type(4)))  float    v4f;

// ---------------------------------------------------------------------------
// Async global->LDS copy of 16 bytes (CDNA5 GLOBAL_LOAD_ASYNC_TO_LDS_B128,
// GV addressing mode: LDS byte address in VGPR, 64-bit global address in
// VGPR pair, SADDR=off). Tracked by ASYNCcnt.
// ---------------------------------------------------------------------------
__device__ __forceinline__ void async_copy16(_Float16* ldsDst, const _Float16* gsrc) {
    unsigned lds = (unsigned)(uintptr_t)ldsDst;   // generic LDS addr truncates to offset
    asm volatile("global_load_async_to_lds_b128 %0, %1, off"
                 :: "v"(lds), "v"(gsrc)
                 : "memory");
}
__device__ __forceinline__ void async_wait0() {
    asm volatile("s_wait_asynccnt 0x0" ::: "memory");
}

// ---------------------------------------------------------------------------
// Pack a row-major f32 [K,N=128] weight matrix into WMMA B-fragment order
// (f16). frag(kb,nb) for lane L is v16h at ((kb*8+nb)*32 + L)*16; half h =
// B[kb*32 + 8*(L>>4) + h + 8*(h>>3)][nb*16 + (L&15)].
// ---------------------------------------------------------------------------
__global__ void pack_b_kernel(const float* __restrict__ B, _Float16* __restrict__ Bp,
                              int K, int N) {
    int idx = blockIdx.x * blockDim.x + threadIdx.x;
    int total = K * N;
    if (idx >= total) return;
    int h    = idx & 15;
    int lane = (idx >> 4) & 31;
    int rest = idx >> 9;
    int NB   = N >> 4;
    int nb   = rest % NB;
    int kb   = rest / NB;
    int g    = lane >> 4;
    int n    = lane & 15;
    int k    = kb * 32 + 8 * g + h + 8 * (h >> 3);
    Bp[idx] = (_Float16)B[(size_t)k * N + nb * 16 + n];
}

// ---------------------------------------------------------------------------
// Load one 16x32 f16 A-fragment from a row-major f32 matrix (lda = K).
// ISA layout: lane L, m = L&15, g = L>>4; halves 0..7 -> k0+8g+h,
// halves 8..15 -> k0+16+8g+(h-8). Two contiguous 8-float runs per lane.
// ---------------------------------------------------------------------------
__device__ __forceinline__ v16h load_a_frag_f32(const float* __restrict__ A, int lda,
                                                int row, int k0, int g) {
    const float* p = A + (size_t)row * lda + k0 + 8 * g;
    v4f a0 = *(const v4f*)(p + 0);
    v4f a1 = *(const v4f*)(p + 4);
    v4f a2 = *(const v4f*)(p + 16);
    v4f a3 = *(const v4f*)(p + 20);
    v16h r;
    r[0]  = (_Float16)a0[0]; r[1]  = (_Float16)a0[1]; r[2]  = (_Float16)a0[2]; r[3]  = (_Float16)a0[3];
    r[4]  = (_Float16)a1[0]; r[5]  = (_Float16)a1[1]; r[6]  = (_Float16)a1[2]; r[7]  = (_Float16)a1[3];
    r[8]  = (_Float16)a2[0]; r[9]  = (_Float16)a2[1]; r[10] = (_Float16)a2[2]; r[11] = (_Float16)a2[3];
    r[12] = (_Float16)a3[0]; r[13] = (_Float16)a3[1]; r[14] = (_Float16)a3[2]; r[15] = (_Float16)a3[3];
    return r;
}

// ---------------------------------------------------------------------------
// Dual-input WMMA GEMM, N fixed to 128:
//   C[M,128] = A1[M,K1]@B1 + (A2 ? A2[M,K2]@B2 : 0) + bias, optional ReLU.
// Block = 256 threads = 8 waves: 4 row-groups x 2 column halves; block tile
// is 64 rows x 128 cols. Packed B fragments are async-staged into LDS once
// per block. Per-wave state: 4 v8f accumulators (32 VGPRs) -> no spills.
// In-place C==A1 is safe (each block only reads/writes its own rows).
// ---------------------------------------------------------------------------
__global__ __launch_bounds__(256, 1)
void wmma_gemm_dual(const float* __restrict__ A1, const _Float16* __restrict__ B1,
                    const float* __restrict__ A2, const _Float16* __restrict__ B2,
                    const float* __restrict__ bias, float* __restrict__ C,
                    int M, int K1, int K2, int do_relu) {
    extern __shared__ _Float16 smem[];
    const int tid     = threadIdx.x;
    const int lane    = tid & 31;
    const int wave    = tid >> 5;
    const int rowgrp  = wave >> 1;
    const int colhalf = wave & 1;
    const int row0    = blockIdx.x * 64 + rowgrp * 16;
    const int g       = lane >> 4;
    const int n       = lane & 15;

    _Float16* sB1 = smem;
    _Float16* sB2 = smem + (size_t)K1 * H;

    // ---- async-stage packed B fragments into LDS (ASYNCcnt) --------------
    {
        int chunks = (K1 * H) >> 3;               // 16-byte chunks
        for (int c = tid; c < chunks; c += 256)
            async_copy16(sB1 + c * 8, B1 + c * 8);
        if (A2) {
            int chunks2 = (K2 * H) >> 3;
            for (int c = tid; c < chunks2; c += 256)
                async_copy16(sB2 + c * 8, B2 + c * 8);
        }
        async_wait0();          // my wave's copies landed in LDS
        __syncthreads();        // everyone's copies landed
    }

    int arow = row0 + n;
    if (arow > M - 1) arow = M - 1;   // clamp OOB rows; stores guarded below

    v8f acc[4];
    #pragma unroll
    for (int i = 0; i < 4; ++i)
        #pragma unroll
        for (int j = 0; j < 8; ++j) acc[i][j] = 0.0f;

    const v16h* sB1v = (const v16h*)sB1;
    const int KB1 = K1 >> 5;
    for (int kb = 0; kb < KB1; ++kb) {
        if (kb + 1 < KB1)
            __builtin_prefetch(A1 + (size_t)arow * K1 + (kb + 1) * 32 + 8 * g, 0, 3);
        v16h a = load_a_frag_f32(A1, K1, arow, kb * 32, g);
        #pragma unroll
        for (int nb = 0; nb < 4; ++nb) {
            v16h b = sB1v[(kb * 8 + colhalf * 4 + nb) * 32 + lane];
            acc[nb] = __builtin_amdgcn_wmma_f32_16x16x32_f16(
                false, a, false, b, (short)0, acc[nb], false, false);
        }
    }
    if (A2) {
        const v16h* sB2v = (const v16h*)sB2;
        const int KB2 = K2 >> 5;
        for (int kb = 0; kb < KB2; ++kb) {
            if (kb + 1 < KB2)
                __builtin_prefetch(A2 + (size_t)arow * K2 + (kb + 1) * 32 + 8 * g, 0, 3);
            v16h a = load_a_frag_f32(A2, K2, arow, kb * 32, g);
            #pragma unroll
            for (int nb = 0; nb < 4; ++nb) {
                v16h b = sB2v[(kb * 8 + colhalf * 4 + nb) * 32 + lane];
                acc[nb] = __builtin_amdgcn_wmma_f32_16x16x32_f16(
                    false, a, false, b, (short)0, acc[nb], false, false);
            }
        }
    }

    // C/D layout: VGPR i -> row row0 + i + 8*g, col = (colhalf*4+nb)*16 + n
    #pragma unroll
    for (int nb = 0; nb < 4; ++nb) {
        int col = (colhalf * 4 + nb) * 16 + n;
        float bv = bias ? bias[col] : 0.0f;
        #pragma unroll
        for (int i = 0; i < 8; ++i) {
            int r = row0 + 8 * g + i;
            if (r < M) {
                float v = acc[nb][i] + bv;
                if (do_relu) v = fmaxf(v, 0.0f);
                C[(size_t)r * H + col] = v;
            }
        }
    }
}

// ---------------------------------------------------------------------------
// Edge gate MLP: 11 -> 32 -> 16 -> 1 per edge, weights staged in LDS.
// ---------------------------------------------------------------------------
__global__ void edge_gate_kernel(const float* __restrict__ ea,
                                 const float* __restrict__ eW1, const float* __restrict__ eb1,
                                 const float* __restrict__ eW2, const float* __restrict__ eb2,
                                 const float* __restrict__ eW3, const float* __restrict__ eb3,
                                 float* __restrict__ wout, int E) {
    __shared__ float sW1[11 * 32], sB1[32], sW2[32 * 16], sB2[16], sW3[16], sB3[1];
    for (int i = threadIdx.x; i < 11 * 32; i += blockDim.x) sW1[i] = eW1[i];
    for (int i = threadIdx.x; i < 32 * 16; i += blockDim.x) sW2[i] = eW2[i];
    for (int i = threadIdx.x; i < 32; i += blockDim.x) sB1[i] = eb1[i];
    for (int i = threadIdx.x; i < 16; i += blockDim.x) { sB2[i] = eb2[i]; sW3[i] = eW3[i]; }
    if (threadIdx.x == 0) sB3[0] = eb3[0];
    __syncthreads();

    int e = blockIdx.x * blockDim.x + threadIdx.x;
    if (e >= E) return;

    float sent  = fminf(fmaxf(ea[2 * e + 0], -1.0f), 1.0f);
    float decay = fmaxf(ea[2 * e + 1], 1e-6f);
    float dt    = fmaxf(-logf(decay) * 100.0f, 0.0f);         // /LAM, LAM=0.01
    float dss   = fminf(fmaxf(dt * (1.0f / 30.0f), 0.0f), 12.0f);

    float x[11];
    x[0] = sent;
    x[1] = log1pf(dt);
    #pragma unroll
    for (int f = 0; f < 4; ++f) {
        x[2 + f] = sinf(dss * (float)(f + 1));
        x[6 + f] = cosf(dss * (float)(f + 1));
    }
    x[10] = dss;

    float h1[32];
    #pragma unroll 4
    for (int j = 0; j < 32; ++j) {
        float s = sB1[j];
        #pragma unroll
        for (int i = 0; i < 11; ++i) s += x[i] * sW1[i * 32 + j];
        h1[j] = fmaxf(s, 0.0f);
    }
    float h2[16];
    #pragma unroll 4
    for (int j = 0; j < 16; ++j) {
        float s = sB2[j];
        #pragma unroll 8
        for (int i = 0; i < 32; ++i) s += h1[i] * sW2[i * 16 + j];
        h2[j] = fmaxf(s, 0.0f);
    }
    float o = sB3[0];
    #pragma unroll
    for (int i = 0; i < 16; ++i) o += h2[i] * sW3[i];
    wout[e] = 1.0f / (1.0f + expf(-o));
}

// ---------------------------------------------------------------------------
// Edge scatter: agg[dst] += x[src] * w. 32 lanes per edge, 4 floats each
// (global_atomic_add_f32).
// ---------------------------------------------------------------------------
__global__ void scatter_edges(const float* __restrict__ x, const int* __restrict__ src,
                              const int* __restrict__ dst, const float* __restrict__ w,
                              float* __restrict__ agg, int E) {
    long long t = (long long)blockIdx.x * blockDim.x + threadIdx.x;
    int e = (int)(t >> 5);
    if (e >= E) return;
    int lane = (int)(t & 31);
    float wt = w[e];
    const v4f xv = *(const v4f*)(x + (size_t)src[e] * H + lane * 4);
    float* ap = agg + (size_t)dst[e] * H + lane * 4;
    atomicAdd(ap + 0, xv[0] * wt);
    atomicAdd(ap + 1, xv[1] * wt);
    atomicAdd(ap + 2, xv[2] * wt);
    atomicAdd(ap + 3, xv[3] * wt);
}

// ---------------------------------------------------------------------------
// x = LayerNorm(relu(nw) + x) * scale + bias, one wave (32 lanes x 4) per row.
// ---------------------------------------------------------------------------
__global__ void relu_add_ln(const float* __restrict__ nw, float* __restrict__ x,
                            const float* __restrict__ sc, const float* __restrict__ bs,
                            int M) {
    int row = blockIdx.x * (blockDim.x >> 5) + (threadIdx.x >> 5);
    if (row >= M) return;
    int lane = threadIdx.x & 31;
    v4f nv = *(const v4f*)(nw + (size_t)row * H + lane * 4);
    v4f pv = *(const v4f*)(x + (size_t)row * H + lane * 4);
    v4f v;
    #pragma unroll
    for (int i = 0; i < 4; ++i) v[i] = fmaxf(nv[i], 0.0f) + pv[i];
    float s  = v[0] + v[1] + v[2] + v[3];
    float ss = v[0] * v[0] + v[1] * v[1] + v[2] * v[2] + v[3] * v[3];
    for (int off = 16; off; off >>= 1) {
        s  += __shfl_xor(s, off, 32);
        ss += __shfl_xor(ss, off, 32);
    }
    float mu   = s * (1.0f / H);
    float var  = ss * (1.0f / H) - mu * mu;
    float rinv = rsqrtf(var + 1e-5f);
    v4f scv = *(const v4f*)(sc + lane * 4);
    v4f bsv = *(const v4f*)(bs + lane * 4);
    v4f o;
    #pragma unroll
    for (int i = 0; i < 4; ++i) o[i] = (v[i] - mu) * rinv * scv[i] + bsv[i];
    *(v4f*)(x + (size_t)row * H + lane * 4) = o;
}

// ---------------------------------------------------------------------------
// Segment-mean pooling: accumulate + counts, then finalize.
// pooled is [64, 256]: cols 0..127 fact mean, 128..255 comp mean.
// ---------------------------------------------------------------------------
__global__ void pool_accum(const float* __restrict__ x, const int* __restrict__ batch,
                           float* __restrict__ pooled, float* __restrict__ counts,
                           int M, int colOff, int cntOff) {
    long long t = (long long)blockIdx.x * blockDim.x + threadIdx.x;
    int row = (int)(t >> 5);
    if (row >= M) return;
    int lane = (int)(t & 31);
    int b = batch[row];
    v4f xv = *(const v4f*)(x + (size_t)row * H + lane * 4);
    float* pp = pooled + (size_t)b * 256 + colOff + lane * 4;
    atomicAdd(pp + 0, xv[0]);
    atomicAdd(pp + 1, xv[1]);
    atomicAdd(pp + 2, xv[2]);
    atomicAdd(pp + 3, xv[3]);
    if (lane == 0) atomicAdd(&counts[cntOff + b], 1.0f);
}

__global__ void pool_finalize(float* __restrict__ pooled, const float* __restrict__ counts) {
    int idx = blockIdx.x * blockDim.x + threadIdx.x;
    if (idx >= BATCH * 256) return;
    int b = idx >> 8;
    int c = idx & 255;
    float cnt = counts[(c >= 128 ? 64 : 0) + b];
    pooled[idx] /= fmaxf(cnt, 1.0f);
}

// ---------------------------------------------------------------------------
// out[b] = hcls[b,:] . Wc2 + bc2   (one wave per row)
// ---------------------------------------------------------------------------
__global__ void cls_out_kernel(const float* __restrict__ hc, const float* __restrict__ Wc2,
                               const float* __restrict__ bc2, float* __restrict__ out) {
    int row = blockIdx.x * (blockDim.x >> 5) + (threadIdx.x >> 5);
    if (row >= BATCH) return;
    int lane = threadIdx.x & 31;
    v4f h = *(const v4f*)(hc + (size_t)row * H + lane * 4);
    v4f w = *(const v4f*)(Wc2 + lane * 4);
    float s = h[0] * w[0] + h[1] * w[1] + h[2] * w[2] + h[3] * w[3];
    for (int off = 16; off; off >>= 1) s += __shfl_xor(s, off, 32);
    if (lane == 0) out[row] = s + bc2[0];
}

__global__ void zero_f(float* __restrict__ p, long long n) {
    long long i = (long long)blockIdx.x * blockDim.x + threadIdx.x;
    long long stride = (long long)gridDim.x * blockDim.x;
    for (; i < n; i += stride) p[i] = 0.0f;
}

// ---------------------------------------------------------------------------
extern "C" void kernel_launch(void* const* d_in, const int* in_sizes, int n_in,
                              void* d_out, int out_size, void* d_ws, size_t ws_size,
                              hipStream_t stream) {
    const float* x_fact     = (const float*)d_in[0];
    const float* x_comp     = (const float*)d_in[1];
    const float* ea_f2c     = (const float*)d_in[2];
    const float* ea_c2f     = (const float*)d_in[3];
    const int*   ei_f2c_src = (const int*)d_in[4];
    const int*   ei_f2c_dst = (const int*)d_in[5];
    const int*   ei_c2f_src = (const int*)d_in[6];
    const int*   ei_c2f_dst = (const int*)d_in[7];
    const int*   fact_batch = (const int*)d_in[8];
    const int*   comp_batch = (const int*)d_in[9];
    const float* W_in_fact  = (const float*)d_in[10];
    const float* b_in_fact  = (const float*)d_in[11];
    const float* W_in_comp  = (const float*)d_in[12];
    const float* b_in_comp  = (const float*)d_in[13];
    const float* eW1        = (const float*)d_in[14];
    const float* eb1        = (const float*)d_in[15];
    const float* eW2        = (const float*)d_in[16];
    const float* eb2        = (const float*)d_in[17];
    const float* eW3        = (const float*)d_in[18];
    const float* eb3        = (const float*)d_in[19];
    const float* W_rel      = (const float*)d_in[20];
    const float* b_rel      = (const float*)d_in[21];
    const float* W_root     = (const float*)d_in[22];
    const float* ln_scale   = (const float*)d_in[23];
    const float* ln_bias    = (const float*)d_in[24];
    const float* Wc1        = (const float*)d_in[25];
    const float* bc1        = (const float*)d_in[26];
    const float* Wc2        = (const float*)d_in[27];
    const float* bc2        = (const float*)d_in[28];
    float* out = (float*)d_out;

    // ---- carve workspace -------------------------------------------------
    char* ws = (char*)d_ws;
    size_t off = 0;
    auto alloc = [&](size_t bytes) -> void* {
        void* p = ws + off;
        off = (off + bytes + 255) & ~(size_t)255;
        return p;
    };
    float* xf     = (float*)alloc((size_t)NFACT * H * 4);
    float* xc     = (float*)alloc((size_t)NCOMP * H * 4);
    float* aggf   = (float*)alloc((size_t)NFACT * H * 4);   // also holds new_f (in-place)
    float* aggc   = (float*)alloc((size_t)NCOMP * H * 4);   // also holds new_c (in-place)
    float* wf2c   = (float*)alloc((size_t)NEDGE * 4);
    float* wc2f   = (float*)alloc((size_t)NEDGE * 4);
    float* pooled = (float*)alloc((size_t)BATCH * 256 * 4);
    float* counts = (float*)alloc(128 * 4);
    float* hcls   = (float*)alloc((size_t)BATCH * H * 4);
    _Float16* Wifp   = (_Float16*)alloc((size_t)DFACT * H * 2);
    _Float16* Wicp   = (_Float16*)alloc((size_t)DCOMP * H * 2);
    _Float16* Wrelp  = (_Float16*)alloc((size_t)NLAYERS * 2 * H * H * 2);
    _Float16* Wrootp = (_Float16*)alloc((size_t)NLAYERS * 2 * H * H * 2);
    _Float16* Wc1p   = (_Float16*)alloc((size_t)2 * H * H * 2);

    auto packB = [&](const float* W, _Float16* Wp, int K, int N) {
        int total = K * N;
        pack_b_kernel<<<(total + 255) / 256, 256, 0, stream>>>(W, Wp, K, N);
    };

    // ---- pack all weight matrices into WMMA B-fragment layout ------------
    packB(W_in_fact, Wifp, DFACT, H);
    packB(W_in_comp, Wicp, DCOMP, H);
    for (int l = 0; l < NLAYERS; ++l)
        for (int s = 0; s < 2; ++s) {
            packB(W_rel  + (size_t)(l * 2 + s) * H * H, Wrelp  + (size_t)(l * 2 + s) * H * H, H, H);
            packB(W_root + (size_t)(l * 2 + s) * H * H, Wrootp + (size_t)(l * 2 + s) * H * H, H, H);
        }
    packB(Wc1, Wc1p, 2 * H, H);

    // ---- edge gates ------------------------------------------------------
    edge_gate_kernel<<<(NEDGE + 255) / 256, 256, 0, stream>>>(
        ea_f2c, eW1, eb1, eW2, eb2, eW3, eb3, wf2c, NEDGE);
    edge_gate_kernel<<<(NEDGE + 255) / 256, 256, 0, stream>>>(
        ea_c2f, eW1, eb1, eW2, eb2, eW3, eb3, wc2f, NEDGE);

    // ---- input projections (WMMA GEMM + bias + relu) ---------------------
    wmma_gemm_dual<<<(NFACT + 63) / 64, 256, (size_t)DFACT * H * 2, stream>>>(
        x_fact, Wifp, nullptr, nullptr, b_in_fact, xf, NFACT, DFACT, 0, 1);
    wmma_gemm_dual<<<(NCOMP + 63) / 64, 256, (size_t)DCOMP * H * 2, stream>>>(
        x_comp, Wicp, nullptr, nullptr, b_in_comp, xc, NCOMP, DCOMP, 0, 1);

    // ---- message-passing layers ------------------------------------------
    const int scatterBlocks = (int)(((long long)NEDGE * 32 + 255) / 256);
    const size_t layerSh = (size_t)(H + H) * H * 2;   // B1 + B2 in LDS
    for (int l = 0; l < NLAYERS; ++l) {
        zero_f<<<512, 256, 0, stream>>>(aggc, (long long)NCOMP * H);
        zero_f<<<4096, 256, 0, stream>>>(aggf, (long long)NFACT * H);
        scatter_edges<<<scatterBlocks, 256, 0, stream>>>(
            xf, ei_f2c_src, ei_f2c_dst, wf2c, aggc, NEDGE);
        scatter_edges<<<scatterBlocks, 256, 0, stream>>>(
            xc, ei_c2f_src, ei_c2f_dst, wc2f, aggf, NEDGE);
        // new_c = agg_c @ W_rel[l,0] + b_rel[l,0] + xc @ W_root[l,0]   (in-place)
        wmma_gemm_dual<<<(NCOMP + 63) / 64, 256, layerSh, stream>>>(
            aggc, Wrelp + (size_t)(l * 2 + 0) * H * H,
            xc,   Wrootp + (size_t)(l * 2 + 0) * H * H,
            b_rel + (size_t)(l * 2 + 0) * H, aggc, NCOMP, H, H, 0);
        // new_f = agg_f @ W_rel[l,1] + b_rel[l,1] + xf @ W_root[l,1]   (in-place)
        wmma_gemm_dual<<<(NFACT + 63) / 64, 256, layerSh, stream>>>(
            aggf, Wrelp + (size_t)(l * 2 + 1) * H * H,
            xf,   Wrootp + (size_t)(l * 2 + 1) * H * H,
            b_rel + (size_t)(l * 2 + 1) * H, aggf, NFACT, H, H, 0);
        // x = LN(relu(new) + prev) * scale + bias
        relu_add_ln<<<(NFACT + 7) / 8, 256, 0, stream>>>(aggf, xf, ln_scale,     ln_bias,     NFACT);
        relu_add_ln<<<(NCOMP + 7) / 8, 256, 0, stream>>>(aggc, xc, ln_scale + H, ln_bias + H, NCOMP);
    }

    // ---- pooling ---------------------------------------------------------
    zero_f<<<64, 256, 0, stream>>>(pooled, (long long)BATCH * 256);
    zero_f<<<1, 128, 0, stream>>>(counts, 128);
    pool_accum<<<(int)(((long long)NFACT * 32 + 255) / 256), 256, 0, stream>>>(
        xf, fact_batch, pooled, counts, NFACT, 0, 0);
    pool_accum<<<(int)(((long long)NCOMP * 32 + 255) / 256), 256, 0, stream>>>(
        xc, comp_batch, pooled, counts, NCOMP, 128, 64);
    pool_finalize<<<64, 256, 0, stream>>>(pooled, counts);

    // ---- classifier ------------------------------------------------------
    wmma_gemm_dual<<<1, 256, (size_t)2 * H * H * 2, stream>>>(
        pooled, Wc1p, nullptr, nullptr, bc1, hcls, BATCH, 2 * H, 0, 1);
    cls_out_kernel<<<8, 256, 0, stream>>>(hcls, Wc2, bc2, out);
}
